// BertAdapterAttentionMask_71459665871069
// MI455X (gfx1250) — matrix operand order repaired and checked by reading, not compiled, not run on hardware
//
#include <hip/hip_runtime.h>

// ---- problem constants (from setup_inputs) ----
#define NH_    16
#define HD_    64
#define B_     8
#define S_     2048
#define H_     1024
#define A_     512
#define NTASK_ 10
#define ROWS_  32      // rows of (B*S) per block tile (two 16-row M-tiles)
#define SMAX_  400.0f

typedef __attribute__((ext_vector_type(16))) __bf16       v16bf;
typedef __attribute__((ext_vector_type(8)))  float        v8f;
typedef __attribute__((ext_vector_type(4)))  unsigned int u32x4;

__device__ __forceinline__ float sigm(float x)  { return 1.0f / (1.0f + __expf(-x)); }
__device__ __forceinline__ float geluf(float x) { return 0.5f * x * (1.0f + erff(x * 0.70710678118f)); }

// ---- WMMA fragment loaders (16x16x32 bf16, wave32; contiguous k order used for both A and B) ----
// A-matrix 16x32: lane = half*16 + m ; element e -> k = e + 16*half
__device__ __forceinline__ v16bf ld_a_frag(const __bf16* rowp /* row m, +k0 */, int half) {
  union { u32x4 u[2]; v16bf v; } r;
  r.u[0] = *(const u32x4*)(rowp + 16 * half);       // k = 16*half .. 16*half+7
  r.u[1] = *(const u32x4*)(rowp + 16 * half + 8);   // k = 16*half+8 .. 16*half+15
  return r.v;
}
// B-matrix 32x16: lane = half*16 + n ; element e -> k = e + 16*half  (32 contiguous bytes)
__device__ __forceinline__ v16bf ld_b_frag(const __bf16* p) {
  union { u32x4 u[2]; v16bf v; } r;
  const u32x4* q = (const u32x4*)p;
  r.u[0] = q[0]; r.u[1] = q[1];
  return r.v;
}

// out[m, n] = sum_k A[m,k] * Bg[n*ldb + k]; one 8-wave group covers NT*8 col-tiles,
// A-frag loaded once per k-step and reused across the wave's NT B-tiles.
template<int NT, int KDIM>
__device__ __forceinline__ void gemm_bf16(const __bf16* __restrict__ As, int lda,
                                          const __bf16* __restrict__ Bg, int ldb,
                                          int lane, int w8, v8f* acc) {
  const int half = lane >> 4;
  const int ml   = lane & 15;
  #pragma unroll 1
  for (int k0 = 0; k0 < KDIM; k0 += 32) {
    v16bf a = ld_a_frag(As + ml * lda + k0, half);
    #pragma unroll
    for (int i = 0; i < NT; ++i) {
      const int n = (w8 * NT + i) * 16 + ml;
      v16bf b = ld_b_frag(Bg + (size_t)n * ldb + k0 + 16 * half);
      acc[i] = __builtin_amdgcn_wmma_f32_16x16x32_bf16(
          false, a, false, b, (short)0, acc[i], false, false);
    }
  }
}

// ====================== prep kernels (gate + query folding) ======================

__global__ void prep_qvec(const float* __restrict__ etask, const float* __restrict__ q_w,
                          const float* __restrict__ q_b,   const float* __restrict__ equery,
                          const int* tptr, const int* sptr, float* __restrict__ qvec) {
  const int h = blockIdx.x * blockDim.x + threadIdx.x;
  if (h >= H_) return;
  const int t = *tptr; const float sf = (float)(*sptr);
  const float* te = etask + (size_t)t * H_;
  const float* qr = q_w + (size_t)h * H_;
  float acc = 0.0f;
  for (int i = 0; i < H_; ++i) acc += te[i] * qr[i];
  qvec[h] = (acc + q_b[h]) * sigm(sf * equery[(size_t)t * H_ + h]);
}

// Kq[d][hi] = sum_n qvec[n*64+d] * gk[n*64+d] * k_w[n*64+d][hi]   (q and gate folded into K proj)
__global__ void prep_kq(const float* __restrict__ qvec, const float* __restrict__ k_w,
                        const float* __restrict__ ekey, const int* tptr, const int* sptr,
                        __bf16* __restrict__ Kq) {
  const int idx = blockIdx.x * blockDim.x + threadIdx.x;
  if (idx >= HD_ * H_) return;
  const int t = *tptr; const float sf = (float)(*sptr);
  const int d = idx >> 10, hi = idx & (H_ - 1);
  float acc = 0.0f;
  for (int n = 0; n < NH_; ++n) {
    const int h = n * HD_ + d;
    acc += qvec[h] * sigm(sf * ekey[(size_t)t * H_ + h]) * k_w[(size_t)h * H_ + hi];
  }
  Kq[idx] = (__bf16)acc;
}

// sbias[d] = sum_n qvec[n*64+d] * gk[n*64+d] * k_b[n*64+d]
__global__ void prep_sbias(const float* __restrict__ qvec, const float* __restrict__ k_b,
                           const float* __restrict__ ekey, const int* tptr, const int* sptr,
                           float* __restrict__ sbias) {
  const int d = threadIdx.x;
  if (d >= HD_) return;
  const int t = *tptr; const float sf = (float)(*sptr);
  float acc = 0.0f;
  for (int n = 0; n < NH_; ++n) {
    const int h = n * HD_ + d;
    acc += qvec[h] * sigm(sf * ekey[(size_t)t * H_ + h]) * k_b[h];
  }
  sbias[d] = acc;
}

__global__ void prep_vec(const float* __restrict__ efc2, const float* __restrict__ v_b,
                         const float* __restrict__ evalue, const int* tptr, const int* sptr,
                         float* __restrict__ g2, float* __restrict__ vbg) {
  const int idx = blockIdx.x * blockDim.x + threadIdx.x;
  const int t = *tptr; const float sf = (float)(*sptr);
  if (idx < NTASK_ * H_) {
    const int j = idx / H_, h = idx % H_;
    const float e = (j == 0) ? sf * efc2[(size_t)t * H_ + h]
                             : SMAX_ * efc2[(size_t)(j - 1) * H_ + h];
    g2[idx] = sigm(e);
  }
  if (idx < H_) vbg[idx] = v_b[idx] * sigm(sf * evalue[(size_t)t * H_ + idx]);
}

__global__ void prep_w1(const float* __restrict__ fc1_w, __bf16* __restrict__ W1) {
  const int idx = blockIdx.x * blockDim.x + threadIdx.x;
  if (idx < A_ * H_) W1[idx] = (__bf16)fc1_w[idx];
}

// W2[j][h][a] = fc2_w[h][a] * sigmoid(gate_j[a])
__global__ void prep_w2(const float* __restrict__ fc2_w, const float* __restrict__ efc1,
                        const int* tptr, const int* sptr, __bf16* __restrict__ W2) {
  const size_t idx = (size_t)blockIdx.x * blockDim.x + threadIdx.x;
  if (idx >= (size_t)NTASK_ * H_ * A_) return;
  const int t = *tptr; const float sf = (float)(*sptr);
  const int j = (int)(idx / (H_ * A_));
  const int r = (int)(idx % (H_ * A_));
  const int a = r % A_;
  const float e = (j == 0) ? sf * efc1[(size_t)t * A_ + a]
                           : SMAX_ * efc1[(size_t)(j - 1) * A_ + a];
  W2[idx] = (__bf16)(fc2_w[r] * sigm(e));
}

// Vw[ho][hi] = v_w[ho][hi] * sigmoid(s*evalue[t][ho])
__global__ void prep_vw(const float* __restrict__ v_w, const float* __restrict__ evalue,
                        const int* tptr, const int* sptr, __bf16* __restrict__ Vw) {
  const int idx = blockIdx.x * blockDim.x + threadIdx.x;
  if (idx >= H_ * H_) return;
  const int t = *tptr; const float sf = (float)(*sptr);
  const int ho = idx / H_;
  Vw[idx] = (__bf16)(v_w[idx] * sigm(sf * evalue[(size_t)t * H_ + ho]));
}

// ====================== fused main kernel ======================
// 512 threads = 16 wave32 waves (two 8-wave groups, one per 16-row M-tile).
// Dynamic LDS: Xs bf16[32*1024] 64K | H1s bf16[32*512] 32K | Hs bf16[32*1024] 64K
//              Vs bf16[32*1024] 64K | scp f32[2][32*64] 16K  => 240 KB < 320 KB WGP LDS
__global__ __launch_bounds__(512)
void fused_main(const float* __restrict__ x,
                const float* __restrict__ b1, const float* __restrict__ b2,
                const __bf16* __restrict__ W1, const __bf16* __restrict__ W2,
                const __bf16* __restrict__ Vw, const __bf16* __restrict__ Kq,
                const float* __restrict__ sbias, const float* __restrict__ g2,
                const float* __restrict__ vbg, const int* __restrict__ tptr,
                float* __restrict__ out) {
  extern __shared__ __align__(16) char smem[];
  __bf16* Xs  = (__bf16*)smem;                       // [32][1024]
  __bf16* H1s = Xs  + ROWS_ * H_;                    // [32][512]
  __bf16* Hs  = H1s + ROWS_ * A_;                    // [32][1024]
  __bf16* Vs  = Hs  + ROWS_ * H_;                    // [32][1024]
  float*  scp = (float*)(Vs + ROWS_ * H_);           // [2][32*64]

  const int tid  = threadIdx.x;
  const int lane = tid & 31, wave = tid >> 5;
  const int half = lane >> 4, nl = lane & 15, ml = lane & 15;
  const int w8 = wave & 7, mtile = wave >> 3, mrow0 = mtile * 16;
  const int row0 = blockIdx.x * ROWS_;
  const int t    = *tptr;

  // stage X tile -> bf16 LDS
  for (int i = tid; i < ROWS_ * H_; i += 512)
    Xs[i] = (__bf16)x[(size_t)(row0 + (i >> 10)) * H_ + (i & (H_ - 1))];
  __syncthreads();

  // GEMM1: H1s = gelu(X @ W1^T + b1)   [32 x 512]
  {
    v8f acc[4];
    #pragma unroll
    for (int i = 0; i < 4; ++i) acc[i] = (v8f)0.0f;
    gemm_bf16<4, H_>(Xs + mrow0 * H_, H_, W1, H_, lane, w8, acc);
    #pragma unroll
    for (int i = 0; i < 4; ++i) {
      const int col = (w8 * 4 + i) * 16 + nl;
      #pragma unroll
      for (int r = 0; r < 8; ++r)
        H1s[(mrow0 + r + 8 * half) * A_ + col] = (__bf16)geluf(acc[i][r] + b1[col]);
    }
  }
  __syncthreads();

  // online-softmax state: 2048 (m,d) pairs / 512 threads = 4 per thread; ctx in regs
  float runmax[4], runsum[4], ctx[4][16];
  #pragma unroll
  for (int i = 0; i < 4; ++i) {
    runmax[i] = -1e30f; runsum[i] = 0.0f;
    #pragma unroll
    for (int n = 0; n < 16; ++n) ctx[i][n] = 0.0f;
  }

  for (int j = 0; j <= t; ++j) {
    // GEMM2: Hs = gelu(H1s @ W2g[j]^T + b2) * g2[j]   [32 x 1024]
    {
      v8f acc[8];
      #pragma unroll
      for (int i = 0; i < 8; ++i) acc[i] = (v8f)0.0f;
      gemm_bf16<8, A_>(H1s + mrow0 * A_, A_, W2 + (size_t)j * H_ * A_, A_, lane, w8, acc);
      #pragma unroll
      for (int i = 0; i < 8; ++i) {
        const int col = (w8 * 8 + i) * 16 + nl;
        const float gg = g2[j * H_ + col];
        #pragma unroll
        for (int r = 0; r < 8; ++r)
          Hs[(mrow0 + r + 8 * half) * H_ + col] = (__bf16)(geluf(acc[i][r] + b2[col]) * gg);
      }
    }
    __syncthreads();

    // V = Hs @ Vwg^T + vbg  -> bf16 LDS   [32 x 1024]
    {
      v8f acc[8];
      #pragma unroll
      for (int i = 0; i < 8; ++i) acc[i] = (v8f)0.0f;
      gemm_bf16<8, H_>(Hs + mrow0 * H_, H_, Vw, H_, lane, w8, acc);
      #pragma unroll
      for (int i = 0; i < 8; ++i) {
        const int col = (w8 * 8 + i) * 16 + nl;
        #pragma unroll
        for (int r = 0; r < 8; ++r)
          Vs[(mrow0 + r + 8 * half) * H_ + col] = (__bf16)(acc[i][r] + vbg[col]);
      }
    }

    // scores = Hs @ Kq^T  [32 x 64]; 8 tiles, K split in half across the 8-wave group:
    // wave-in-group: ntile = w8&3, khalf = w8>>2 -> partial sums into scp[khalf]
    {
      const int ntile = w8 & 3, kh = w8 >> 2;
      const int kbase = kh * (H_ / 2);
      v8f acc = (v8f)0.0f;
      const __bf16* As = Hs + mrow0 * H_;
      #pragma unroll 1
      for (int k0 = 0; k0 < H_ / 2; k0 += 32) {
        v16bf a = ld_a_frag(As + ml * H_ + kbase + k0, half);
        v16bf b = ld_b_frag(Kq + (size_t)(ntile * 16 + ml) * H_ + kbase + k0 + 16 * half);
        acc = __builtin_amdgcn_wmma_f32_16x16x32_bf16(
            false, a, false, b, (short)0, acc, false, false);
      }
      #pragma unroll
      for (int r = 0; r < 8; ++r)
        scp[kh * (ROWS_ * HD_) + (mrow0 + r + 8 * half) * HD_ + ntile * 16 + nl] = acc[r];
    }
    __syncthreads();

    // online softmax over tasks
    #pragma unroll
    for (int i = 0; i < 4; ++i) {
      const int p = tid * 4 + i, m = p >> 6, d = p & 63;
      const float sc = (scp[m * HD_ + d] + scp[ROWS_ * HD_ + m * HD_ + d] + sbias[d]) * 0.125f;
      const float mnew = fmaxf(runmax[i], sc);
      const float corr = __expf(runmax[i] - mnew);
      const float w    = __expf(sc - mnew);
      runsum[i] = runsum[i] * corr + w;
      #pragma unroll
      for (int n = 0; n < 16; ++n)
        ctx[i][n] = ctx[i][n] * corr + w * (float)Vs[m * H_ + n * HD_ + d];
      runmax[i] = mnew;
    }
    __syncthreads();
  }

  // out[row, d*16 + n] = x + ctx/runsum   (16 contiguous floats per pair)
  #pragma unroll
  for (int i = 0; i < 4; ++i) {
    const int p = tid * 4 + i, m = p >> 6, d = p & 63;
    const size_t base = (size_t)(row0 + m) * H_ + d * NH_;
    const float inv = 1.0f / runsum[i];
    #pragma unroll
    for (int n = 0; n < 16; ++n) out[base + n] = x[base + n] + ctx[i][n] * inv;
  }
}

// ====================== host launcher ======================
extern "C" void kernel_launch(void* const* d_in, const int* in_sizes, int n_in,
                              void* d_out, int out_size, void* d_ws, size_t ws_size,
                              hipStream_t stream) {
  const float* x      = (const float*)d_in[0];
  const float* fc1_w  = (const float*)d_in[1];
  const float* fc1_b  = (const float*)d_in[2];
  const float* fc2_w  = (const float*)d_in[3];
  const float* fc2_b  = (const float*)d_in[4];
  const float* efc1   = (const float*)d_in[5];
  const float* efc2   = (const float*)d_in[6];
  const float* etask  = (const float*)d_in[7];
  const float* q_w    = (const float*)d_in[8];
  const float* q_b    = (const float*)d_in[9];
  const float* k_w    = (const float*)d_in[10];
  const float* k_b    = (const float*)d_in[11];
  const float* v_w    = (const float*)d_in[12];
  const float* v_b    = (const float*)d_in[13];
  const float* equery = (const float*)d_in[14];
  const float* ekey   = (const float*)d_in[15];
  const float* evalue = (const float*)d_in[16];
  const int*   tptr   = (const int*)d_in[17];
  const int*   sptr   = (const int*)d_in[18];
  float* out = (float*)d_out;

  char* ws = (char*)d_ws;
  size_t off = 0;
  float*  qvec  = (float*)(ws + off); off += (size_t)H_ * sizeof(float);
  float*  g2    = (float*)(ws + off); off += (size_t)NTASK_ * H_ * sizeof(float);
  float*  vbg   = (float*)(ws + off); off += (size_t)H_ * sizeof(float);
  float*  sbias = (float*)(ws + off); off += 4096;
  __bf16* W1    = (__bf16*)(ws + off); off += (size_t)A_ * H_ * 2;
  __bf16* W2    = (__bf16*)(ws + off); off += (size_t)NTASK_ * H_ * A_ * 2;
  __bf16* Kq    = (__bf16*)(ws + off); off += (size_t)HD_ * H_ * 2;
  __bf16* Vw    = (__bf16*)(ws + off); off += (size_t)H_ * H_ * 2;
  (void)off; (void)ws_size; (void)in_sizes; (void)n_in; (void)out_size;

  prep_qvec <<<(H_ + 255) / 256, 256, 0, stream>>>(etask, q_w, q_b, equery, tptr, sptr, qvec);
  prep_kq   <<<(HD_ * H_ + 255) / 256, 256, 0, stream>>>(qvec, k_w, ekey, tptr, sptr, Kq);
  prep_sbias<<<1, 64, 0, stream>>>(qvec, k_b, ekey, tptr, sptr, sbias);
  prep_vec  <<<(NTASK_ * H_ + 255) / 256, 256, 0, stream>>>(efc2, v_b, evalue, tptr, sptr, g2, vbg);
  prep_w1   <<<(A_ * H_ + 255) / 256, 256, 0, stream>>>(fc1_w, W1);
  prep_w2   <<<(int)(((size_t)NTASK_ * H_ * A_ + 255) / 256), 256, 0, stream>>>(fc2_w, efc1,
                                                                                tptr, sptr, W2);
  prep_vw   <<<(H_ * H_ + 255) / 256, 256, 0, stream>>>(v_w, evalue, tptr, sptr, Vw);

  const size_t shmem = (size_t)ROWS_ * H_ * 2   // Xs  bf16
                     + (size_t)ROWS_ * A_ * 2   // H1s bf16
                     + (size_t)ROWS_ * H_ * 2   // Hs  bf16
                     + (size_t)ROWS_ * H_ * 2   // Vs  bf16
                     + 2 * (size_t)ROWS_ * HD_ * 4; // scp f32 = 245760 B
  fused_main<<<(B_ * S_) / ROWS_, 512, shmem, stream>>>(x, fc1_b, fc2_b, W1, W2, Vw, Kq,
                                                        sbias, g2, vbg, tptr, out);
}
